// HGNNExpertCoupler_7060926234638
// MI455X (gfx1250) — matrix operand order repaired and checked by reading, not compiled, not run on hardware
//
#include <hip/hip_runtime.h>
#include <hip/hip_bf16.h>
#include <math.h>

// ---- problem constants (match reference) ----
constexpr int Bb   = 4;
constexpr int Ll   = 1024;
constexpr int G    = Bb * Ll;     // 4096 tokens
constexpr int E    = 8;           // experts / hypergraph nodes per token
constexpr int D    = 512;         // feature dim
constexpr int NL   = 2;           // layers
constexpr int TOK  = 16;          // tokens per workgroup
constexpr int ROWS = TOK * E;     // 128 LDS rows per workgroup
constexpr int LDH  = D + 8;       // padded LDS row stride (elements); 520*2B = 16B-aligned rows

typedef __bf16 v16bf __attribute__((ext_vector_type(16)));
typedef float  v8f   __attribute__((ext_vector_type(8)));

// ws layout (bytes)
constexpr size_t WT_BYTES   = (size_t)D * D * sizeof(__bf16);  // 512 KB per matrix
constexpr size_t WS_M_OFF   = 3 * WT_BYTES;                    // 8x8 mix matrix (f32)

// -------------------------------------------------------------------------
// Setup kernel 1: convert lin_w[0], lin_w[1], comb_w (f32, [d][f]) into
// transposed bf16  Wt[f][d]  so WMMA B-tiles are contiguous per lane.
// -------------------------------------------------------------------------
__global__ void prep_weights(const float* __restrict__ lw,   // (2, D, D)
                             const float* __restrict__ cw,   // (D, D)
                             __bf16* __restrict__ wt)        // 3 * D * D
{
    int idx = blockIdx.x * blockDim.x + threadIdx.x;   // < 3*D*D = 786432
    int m   = idx >> 18;            // which matrix (D*D = 2^18)
    int i   = idx & (D * D - 1);
    int dd  = i >> 9;               // source row  (d)
    int f   = i & (D - 1);          // source col  (f)
    const float* src = (m < 2) ? (lw + (size_t)m * D * D) : cw;
    wt[(size_t)m * D * D + (size_t)f * D + dd] = (__bf16)src[(size_t)dd * D + f];
}

// -------------------------------------------------------------------------
// Setup kernel 2: build the 8x8 hypergraph mixing matrix
//   M = D^{-1} H B^{-1} H^T   from hyperedge_index (2, nnz).
// -------------------------------------------------------------------------
__global__ void prep_M(const int* __restrict__ he, int nnz, float* __restrict__ Mm)
{
    int t = threadIdx.x;
    if (t >= E * E) return;
    int i = t >> 3, j = t & 7;
    const int* node = he;
    const int* edge = he + nnz;
    float degn = 0.f;
    for (int p = 0; p < nnz; ++p) if (node[p] == i) degn += 1.f;
    float acc = 0.f;
    for (int p = 0; p < nnz; ++p) {
        if (node[p] != i) continue;
        int e = edge[p];
        float dege = 0.f;
        for (int q = 0; q < nnz; ++q) if (edge[q] == e) dege += 1.f;
        float binv = (dege > 0.f) ? 1.f / dege : 0.f;
        for (int q = 0; q < nnz; ++q) if (edge[q] == e && node[q] == j) acc += binv;
    }
    Mm[t] = (degn > 0.f) ? acc / degn : 0.f;
}

// -------------------------------------------------------------------------
// One 16x16 output tile of  A(16xD, bf16 LDS, stride LDH) @ B(DxD via
// transposed bf16 Wt[f][d], stride D), f32 accumulate, K swept by 16
// v_wmma_f32_16x16x32_bf16 steps.
//
// A layout (16-bit A 16x32, same K order as B): lane = M row; lanes<16 take K
// 0..15, lanes>=16 take K 16..31  -> one
// contiguous 32B run per lane.
// B layout (16-bit B 32x16): lane = N col; lanes<16 take K 0..15,
// lanes>=16 take K 16..31 -> one contiguous 32B run per lane of Wt row n.
// -------------------------------------------------------------------------
__device__ inline v8f gemm_tile(const __bf16* __restrict__ A,
                                const __bf16* __restrict__ Bt,
                                int rb, int cb, int lane)
{
    const int m   = lane & 15;
    const int kbA = (lane < 16) ? 0 : 16;
    const int n   = cb * 16 + (lane & 15);
    const int kbB = (lane < 16) ? 0 : 16;
    const __bf16* arow = A  + (size_t)(rb * 16 + m) * LDH;
    const __bf16* brow = Bt + (size_t)n * D;
    v8f acc = {};
#pragma unroll 4
    for (int kt = 0; kt < D; kt += 32) {
        union { v16bf v; uint4 q[2]; } ua, ub;
        ua.q[0] = *(const uint4*)(arow + kt + kbA);
        ua.q[1] = *(const uint4*)(arow + kt + kbA + 8);
        ub.q[0] = *(const uint4*)(brow + kt + kbB);
        ub.q[1] = *(const uint4*)(brow + kt + kbB + 8);
        acc = __builtin_amdgcn_wmma_f32_16x16x32_bf16(
                  false, ua.v, false, ub.v, (short)0, acc, false, false);
    }
    return acc;
}

// -------------------------------------------------------------------------
// Fused: stage -> [GEMM -> 8x8 mix + bias] x2 -> expert mean -> combiner
// GEMM -> exact GELU -> LayerNorm -> store. All activations LDS-resident.
// -------------------------------------------------------------------------
__global__ __launch_bounds__(256, 1)
void hgnn_fused(const float* __restrict__ x,      // (G, E, D)
                const __bf16* __restrict__ wt0,
                const __bf16* __restrict__ wt1,
                const __bf16* __restrict__ wtc,
                const float* __restrict__ lin_b,  // (2, D)
                const float* __restrict__ comb_b, // (D)
                const float* __restrict__ ln_g,   // (D)
                const float* __restrict__ ln_bt,  // (D)
                const float* __restrict__ Mmat,   // (E*E)
                float* __restrict__ out)          // (G, D)
{
    extern __shared__ char smem[];
    __bf16* Hs = (__bf16*)smem;                                   // 128 x LDH bf16
    __bf16* Ts = (__bf16*)(smem + (size_t)ROWS * LDH * 2);        // 128 x LDH bf16
    float*  Ms = (float*)(smem + (size_t)ROWS * LDH * 4);         // 64 f32

    const int tid  = threadIdx.x;
    const int lane = tid & 31;
    const int wave = tid >> 5;
    const int g0   = blockIdx.x * TOK;

    if (tid < E * E) Ms[tid] = Mmat[tid];

    // ---- stage input tile: f32 -> bf16 LDS ----
    const float* src = x + (size_t)g0 * E * D;
    for (int i = tid; i < ROWS * (D / 4); i += 256) {
        int r  = i / (D / 4);
        int c4 = (i % (D / 4)) * 4;
        float4 v = *(const float4*)(src + (size_t)r * D + c4);
        __bf16* dst = Hs + (size_t)r * LDH + c4;
        dst[0] = (__bf16)v.x; dst[1] = (__bf16)v.y;
        dst[2] = (__bf16)v.z; dst[3] = (__bf16)v.w;
    }
    __syncthreads();

    // ---- two hypergraph-conv layers ----
    for (int li = 0; li < NL; ++li) {
        const __bf16* Wt = li ? wt1 : wt0;
        // wave w owns row-block w (16 rows), all 32 col-blocks
        for (int cb = 0; cb < D / 16; ++cb) {
            v8f acc = gemm_tile(Hs, Wt, wave, cb, lane);
            int n  = cb * 16 + (lane & 15);
            int mb = wave * 16 + ((lane < 16) ? 0 : 8);
#pragma unroll
            for (int r = 0; r < 8; ++r)
                Ts[(size_t)(mb + r) * LDH + n] = (__bf16)acc[r];
        }
        __syncthreads();

        // 8x8 expert mix + bias:  Hs = M * Ts + b
        const float* lb = lin_b + li * D;
        for (int i = tid; i < TOK * D; i += 256) {
            int c  = i & (D - 1);
            int tt = i >> 9;
            float te[E];
#pragma unroll
            for (int e = 0; e < E; ++e)
                te[e] = (float)Ts[(size_t)(tt * E + e) * LDH + c];
            float bias = lb[c];
#pragma unroll
            for (int e = 0; e < E; ++e) {
                float s = bias;
#pragma unroll
                for (int e2 = 0; e2 < E; ++e2) s += Ms[e * E + e2] * te[e2];
                Hs[(size_t)(tt * E + e) * LDH + c] = (__bf16)s;
            }
        }
        __syncthreads();
    }

    // ---- expert mean -> coord (16 x D) in Ts ----
    __bf16* Cs = Ts;
    for (int i = tid; i < TOK * D; i += 256) {
        int c  = i & (D - 1);
        int tt = i >> 9;
        float s = 0.f;
#pragma unroll
        for (int e = 0; e < E; ++e)
            s += (float)Hs[(size_t)(tt * E + e) * LDH + c];
        Cs[(size_t)tt * LDH + c] = (__bf16)(s * (1.f / (float)E));
    }
    __syncthreads();

    // ---- combiner GEMM (16 x D) + bias + exact GELU, f32 into reused LDS ----
    float* Yv = (float*)smem;   // 16 x LDH f32 (aliases Hs region, now dead)
    for (int t4 = 0; t4 < 4; ++t4) {
        int cb = wave + t4 * 8;
        v8f acc = gemm_tile(Cs, wtc, 0, cb, lane);
        int n  = cb * 16 + (lane & 15);
        float cbias = comb_b[n];
        int mb = (lane < 16) ? 0 : 8;
#pragma unroll
        for (int r = 0; r < 8; ++r) {
            float yv = acc[r] + cbias;
            yv = 0.5f * yv * (1.0f + erff(yv * 0.7071067811865476f));
            Yv[(size_t)(mb + r) * LDH + n] = yv;
        }
    }
    __syncthreads();

    // ---- LayerNorm per token row (wave32 shuffle reduction) ----
    for (int rr = 0; rr < 2; ++rr) {
        int row = wave * 2 + rr;                 // token within tile
        float s = 0.f, ss = 0.f;
        for (int c = lane; c < D; c += 32) {
            float v = Yv[(size_t)row * LDH + c];
            s += v; ss += v * v;
        }
#pragma unroll
        for (int off = 16; off > 0; off >>= 1) {
            s  += __shfl_xor(s,  off, 32);
            ss += __shfl_xor(ss, off, 32);
        }
        float mu  = s * (1.f / (float)D);
        float var = ss * (1.f / (float)D) - mu * mu;
        float inv = rsqrtf(var + 1e-5f);
        float* orow = out + (size_t)(g0 + row) * D;
        for (int c = lane; c < D; c += 32)
            orow[c] = (Yv[(size_t)row * LDH + c] - mu) * inv * ln_g[c] + ln_bt[c];
    }
}

// -------------------------------------------------------------------------
extern "C" void kernel_launch(void* const* d_in, const int* in_sizes, int n_in,
                              void* d_out, int out_size, void* d_ws, size_t ws_size,
                              hipStream_t stream)
{
    const float* x     = (const float*)d_in[0];   // expert_outputs (B,L,E,D)
    const int*   he    = (const int*)  d_in[1];   // hyperedge_index (2, nnz)
    const float* lw    = (const float*)d_in[2];   // lin_w (2,D,D)
    const float* lb    = (const float*)d_in[3];   // lin_b (2,D)
    const float* cw    = (const float*)d_in[4];   // comb_w (D,D)
    const float* cb    = (const float*)d_in[5];   // comb_b (D)
    const float* lg    = (const float*)d_in[6];   // ln_g (D)
    const float* lbeta = (const float*)d_in[7];   // ln_beta (D)

    char*   ws = (char*)d_ws;
    __bf16* wt = (__bf16*)ws;                         // 3 transposed bf16 matrices
    float*  Mm = (float*)(ws + WS_M_OFF);             // 8x8 mix matrix
    int nnz = in_sizes[1] / 2;

    prep_weights<<<(3 * D * D) / 256, 256, 0, stream>>>(lw, cw, wt);
    prep_M<<<1, 64, 0, stream>>>(he, nnz, Mm);

    size_t lds_bytes = (size_t)ROWS * LDH * 2 * sizeof(__bf16) + 64 * sizeof(float);
    hgnn_fused<<<G / TOK, 256, lds_bytes, stream>>>(
        x, wt, wt + (size_t)D * D, wt + (size_t)2 * D * D,
        lb, cb, lg, lbeta, Mm, (float*)d_out);
}
